// AbstractAtt_41257455846061
// MI455X (gfx1250) — compile-verified, matrix operand order checked by reading, not run on hardware
//
#include <hip/hip_runtime.h>
#include <hip/hip_bf16.h>
#include <math.h>

// ---------------- problem constants ----------------
#define B_    64
#define DV_   2048
#define NPOS_ 196          // 14*14
#define DQ_   2400
#define DA_   1200
#define G_    2
#define DH_   1200
#define GDH_  2400         // G*DH
#define NC_   3000
#define MTOT_ (B_ * NPOS_) // 12544, divisible by 16

// ---------------- types ----------------
typedef __attribute__((ext_vector_type(16))) __bf16 v16bf;
typedef __attribute__((ext_vector_type(8)))  float  v8f;

union FragBF { v16bf v; uint4 u[2]; };

__device__ __forceinline__ unsigned short f2bf(float f) {
  unsigned int u = __float_as_uint(f);
  unsigned int r = u + 0x7fffu + ((u >> 16) & 1u);   // round-to-nearest-even
  return (unsigned short)(r >> 16);
}
__device__ __forceinline__ float bf2f(unsigned short s) {
  return __uint_as_float(((unsigned int)s) << 16);
}

// ---------------- CDNA5 async memory->LDS helpers ----------------
// GLOBAL_LOAD_ASYNC_TO_LDS_B128: LDS byte address in a VGPR (vdst slot), 64-bit global
// address in a VGPR pair. Tracked by ASYNCcnt; async loads complete in order.
__device__ __forceinline__ void async_copy_b128(const unsigned short* gptr,
                                                const unsigned short* lptr) {
  unsigned int lds_off = (unsigned int)(unsigned long long)(uintptr_t)lptr;
  unsigned long long ga = (unsigned long long)(uintptr_t)gptr;
  asm volatile("global_load_async_to_lds_b128 %0, %1, off"
               :: "v"(lds_off), "v"(ga)
               : "memory");
}

template <int N>
__device__ __forceinline__ void s_wait_async() {
  asm volatile("s_wait_asynccnt %0" :: "i"(N) : "memory");
}

// ---------------- conversion kernels ----------------
__global__ void k_f32_to_bf16(const float* __restrict__ in,
                              unsigned short* __restrict__ out, long long n) {
  long long i = (long long)blockIdx.x * blockDim.x + threadIdx.x;
  long long stride = (long long)gridDim.x * blockDim.x;
  for (; i < n; i += stride) out[i] = f2bf(in[i]);
}

// input_v [B, DV, NPOS] f32  ->  Xv [B*NPOS, DV] bf16 (transpose per batch)
__global__ void k_transpose_v(const float* __restrict__ v,
                              unsigned short* __restrict__ xv) {
  long long i = (long long)blockIdx.x * blockDim.x + threadIdx.x;
  const long long total = (long long)B_ * DV_ * NPOS_;
  if (i >= total) return;
  int p = (int)(i % NPOS_);
  long long t = i / NPOS_;
  int c = (int)(t % DV_);
  int b = (int)(t / DV_);
  xv[((long long)(b * NPOS_ + p)) * DV_ + c] = f2bf(v[i]);
}

// ---------------- WMMA GEMM: D[M,N] = act(A[M,K] * W[N,K]^T + bias[N]) ----------------
// Block: 256 threads = 8 waves; block tile 128(M) x 128(N); wave tile 32x64 = 2x4 WMMA frags
// (8 WMMAs / K-tile / wave). Double-buffered LDS staged via GLOBAL_LOAD_ASYNC_TO_LDS_B128:
// a uniform 4 async b128 per thread per K-tile (row clamping => no divergence; OOB rows feed
// only output rows/cols that are never stored). Steady-state: issue t+1, s_wait_asynccnt 4
// (tile t landed, in-order completion), barrier, compute from LDS, barrier. Final tile peeled
// so the inner loop has no predicates.
template<bool TANH, bool OUTBF>
__global__ __launch_bounds__(256)
void k_gemm_bf16(const unsigned short* __restrict__ A, int lda,
                 const unsigned short* __restrict__ W, int ldb,
                 const float* __restrict__ bias,
                 void* __restrict__ Cout, int ldc,
                 int M, int N, int K) {
  __shared__ unsigned short sA[2][128 * 32];  // 2 x 8 KB
  __shared__ unsigned short sB[2][128 * 32];  // 2 x 8 KB

  const int tid  = threadIdx.x;
  const int lane = tid & 31;
  const int wid  = tid >> 5;
  const int wm   = wid >> 1;     // 0..3 -> M sub-tile of 32
  const int wn   = wid & 1;      // 0..1 -> N sub-tile of 64
  const int m0   = blockIdx.y * 128;
  const int n0   = blockIdx.x * 128;
  const int lane15 = lane & 15;
  const int khalf  = (lane >> 4) << 3;     // 0 or 8 (ISA 16-bit A/B striping)

  // staging: A tile 128x32 and B tile 128x32, each 512 x 16B segs -> 4 segs/thread total
  const int cs   = (tid & 3) << 3;         // k-column start (halves) within tile
  const int rowS = tid >> 2;               // 0..63
  int mA0 = m0 + rowS;      if (mA0 >= M) mA0 = M - 1;
  int mA1 = m0 + rowS + 64; if (mA1 >= M) mA1 = M - 1;
  int nB0 = n0 + rowS;      if (nB0 >= N) nB0 = N - 1;
  int nB1 = n0 + rowS + 64; if (nB1 >= N) nB1 = N - 1;
  const unsigned short* gA0 = A + (size_t)mA0 * lda + cs;
  const unsigned short* gA1 = A + (size_t)mA1 * lda + cs;
  const unsigned short* gB0 = W + (size_t)nB0 * ldb + cs;
  const unsigned short* gB1 = W + (size_t)nB1 * ldb + cs;

  auto issue = [&](int t, int buf) {
    const int k0 = t << 5;
    async_copy_b128(gA0 + k0, &sA[buf][rowS * 32 + cs]);
    async_copy_b128(gA1 + k0, &sA[buf][(rowS + 64) * 32 + cs]);
    async_copy_b128(gB0 + k0, &sB[buf][rowS * 32 + cs]);
    async_copy_b128(gB1 + k0, &sB[buf][(rowS + 64) * 32 + cs]);
  };

  v8f acc0[4] = {{}, {}, {}, {}};
  v8f acc1[4] = {{}, {}, {}, {}};

  auto compute = [&](int cur) {
    const unsigned short* pA = sA[cur];
    const unsigned short* pB = sB[cur];
    FragBF a0, a1, b[4];
    const int r0 = (wm * 32 + lane15) * 32;
    a0.u[0] = *(const uint4*)&pA[r0 + khalf];
    a0.u[1] = *(const uint4*)&pA[r0 + khalf + 16];
    const int r1 = r0 + 16 * 32;
    a1.u[0] = *(const uint4*)&pA[r1 + khalf];
    a1.u[1] = *(const uint4*)&pA[r1 + khalf + 16];
#pragma unroll
    for (int j = 0; j < 4; ++j) {
      const int cj = (wn * 64 + j * 16 + lane15) * 32;
      b[j].u[0] = *(const uint4*)&pB[cj + khalf];
      b[j].u[1] = *(const uint4*)&pB[cj + khalf + 16];
    }
#pragma unroll
    for (int j = 0; j < 4; ++j) {
      acc0[j] = __builtin_amdgcn_wmma_f32_16x16x32_bf16(false, a0.v, false, b[j].v, (short)0, acc0[j], false, false);
      acc1[j] = __builtin_amdgcn_wmma_f32_16x16x32_bf16(false, a1.v, false, b[j].v, (short)0, acc1[j], false, false);
    }
  };

  const int nt = K >> 5;                   // K multiple of 32; nt >= 2 for all our GEMMs
  issue(0, 0);
  for (int t = 0; t < nt - 1; ++t) {       // steady state: predicate-free body
    issue(t + 1, (t + 1) & 1);
    s_wait_async<4>();                     // in-order completion => tile t landed
    __syncthreads();
    compute(t & 1);
    __syncthreads();                       // buffer t&1 free for tile t+2
  }
  s_wait_async<0>();                       // drain: last tile
  __syncthreads();
  compute((nt - 1) & 1);

  // epilogue: D layout — lanes 0-15: VGPR r -> M=r; lanes 16-31: VGPR r -> M=8+r; N = lane&15
  const int mrow = (lane >> 4) << 3;
  auto store = [&](float val, int m, int n) {
    if (m < M && n < N) {
      float x = val + (bias ? bias[n] : 0.0f);
      if (TANH) x = tanhf(x);
      if (OUTBF)
        ((unsigned short*)Cout)[(size_t)m * ldc + n] = f2bf(x);
      else
        ((float*)Cout)[(size_t)m * ldc + n] = x;
    }
  };
#pragma unroll
  for (int j = 0; j < 4; ++j) {
    const int nj = n0 + wn * 64 + j * 16 + lane15;
#pragma unroll
    for (int r = 0; r < 8; ++r) {
      const int mA = m0 + wm * 32 + mrow + r;
      store(acc0[j][r], mA, nj);
      store(acc1[j][r], mA + 16, nj);
    }
  }
}

// ---------------- attention glue kernels ----------------
// wgt[b,g,p] = sum_d tanh(xv[b*196+p, d] * xq[b, d]) * Watt[g,d] + batt[g]
__global__ __launch_bounds__(256)
void k_wgt(const unsigned short* __restrict__ xv,
           const unsigned short* __restrict__ xq,
           const float* __restrict__ Watt,
           const float* __restrict__ batt,
           float* __restrict__ wgt) {
  int wid  = blockIdx.x * 8 + (threadIdx.x >> 5);
  int lane = threadIdx.x & 31;
  if (wid >= B_ * NPOS_) return;
  int b = wid / NPOS_, p = wid % NPOS_;
  const unsigned short* xvr = xv + (size_t)wid * DA_;
  const unsigned short* xqr = xq + (size_t)b * DA_;
  float a0 = 0.f, a1 = 0.f;
  for (int d = lane; d < DA_; d += 32) {
    float t = tanhf(bf2f(xvr[d]) * bf2f(xqr[d]));
    a0 += t * Watt[d];
    a1 += t * Watt[DA_ + d];
  }
  for (int off = 16; off; off >>= 1) {
    a0 += __shfl_xor(a0, off);
    a1 += __shfl_xor(a1, off);
  }
  if (lane == 0) {
    wgt[(size_t)b * G_ * NPOS_ + p]         = a0 + batt[0];
    wgt[(size_t)b * G_ * NPOS_ + NPOS_ + p] = a1 + batt[1];
  }
}

// softmax over NPOS per (b,g) row
__global__ __launch_bounds__(256)
void k_softmax(const float* __restrict__ wgt, float* __restrict__ att) {
  int wid  = blockIdx.x * 8 + (threadIdx.x >> 5);
  int lane = threadIdx.x & 31;
  if (wid >= B_ * G_) return;
  const float* row = wgt + (size_t)wid * NPOS_;
  float mx = -1e30f;
  for (int p = lane; p < NPOS_; p += 32) mx = fmaxf(mx, row[p]);
  for (int off = 16; off; off >>= 1) mx = fmaxf(mx, __shfl_xor(mx, off));
  float s = 0.f;
  for (int p = lane; p < NPOS_; p += 32) s += __expf(row[p] - mx);
  for (int off = 16; off; off >>= 1) s += __shfl_xor(s, off);
  float inv = 1.0f / s;
  float* orow = att + (size_t)wid * NPOS_;
  for (int p = lane; p < NPOS_; p += 32) orow[p] = __expf(row[p] - mx) * inv;
}

// v_att[b,g,d] = sum_p att[b,g,p] * input_v[b,d,p]  (both glimpses per wave)
__global__ __launch_bounds__(256)
void k_vatt(const float* __restrict__ v,
            const float* __restrict__ att,
            unsigned short* __restrict__ vatt) {
  int wid  = blockIdx.x * 8 + (threadIdx.x >> 5);
  int lane = threadIdx.x & 31;
  if (wid >= B_ * DV_) return;
  int b = wid >> 11;          // / 2048
  int d = wid & (DV_ - 1);
  const float* vr  = v + ((size_t)b * DV_ + d) * NPOS_;
  const float* a0r = att + (size_t)b * G_ * NPOS_;
  const float* a1r = a0r + NPOS_;
  float a0 = 0.f, a1 = 0.f;
  for (int p = lane; p < NPOS_; p += 32) {
    float x = vr[p];
    a0 += a0r[p] * x;
    a1 += a1r[p] * x;
  }
  for (int off = 16; off; off >>= 1) {
    a0 += __shfl_xor(a0, off);
    a1 += __shfl_xor(a1, off);
  }
  if (lane == 0) {
    vatt[((size_t)b * G_ + 0) * DV_ + d] = f2bf(a0);
    vatt[((size_t)b * G_ + 1) * DV_ + d] = f2bf(a1);
  }
}

// elementwise bf16 product
__global__ void k_hadamard_bf(const unsigned short* __restrict__ a,
                              const unsigned short* __restrict__ b,
                              unsigned short* __restrict__ o, int n) {
  int i = blockIdx.x * blockDim.x + threadIdx.x;
  if (i < n) o[i] = f2bf(bf2f(a[i]) * bf2f(b[i]));
}

// ---------------- host launch ----------------
static inline size_t align256(size_t x) { return (x + 255) & ~(size_t)255; }

extern "C" void kernel_launch(void* const* d_in, const int* in_sizes, int n_in,
                              void* d_out, int out_size, void* d_ws, size_t ws_size,
                              hipStream_t stream) {
  const float* input_q = (const float*)d_in[0];
  const float* input_v = (const float*)d_in[1];
  const float* Wv_att  = (const float*)d_in[2];
  const float* bv_att  = (const float*)d_in[3];
  const float* Wq_att  = (const float*)d_in[4];
  const float* bq_att  = (const float*)d_in[5];
  const float* Watt    = (const float*)d_in[6];
  const float* batt    = (const float*)d_in[7];
  const float* Wv_fus  = (const float*)d_in[8];
  const float* bv_fus  = (const float*)d_in[9];
  const float* Wq_fus  = (const float*)d_in[10];
  const float* bq_fus  = (const float*)d_in[11];
  const float* Wc      = (const float*)d_in[12];
  const float* bc      = (const float*)d_in[13];

  float* x_out   = (float*)d_out;                  // [B, NC]
  float* wgt_out = x_out + (size_t)B_ * NC_;       // [B, G, NPOS]

  // workspace carve-out
  size_t off = 0;
  auto carve = [&](size_t bytes) { size_t r = off; off = align256(off + bytes); return r; };
  char* ws = (char*)d_ws;
  unsigned short* Xv_bf     = (unsigned short*)(ws + carve((size_t)MTOT_ * DV_ * 2));
  unsigned short* q_bf      = (unsigned short*)(ws + carve((size_t)B_ * DQ_ * 2));
  unsigned short* Wv_att_bf = (unsigned short*)(ws + carve((size_t)DA_ * DV_ * 2));
  unsigned short* Wq_att_bf = (unsigned short*)(ws + carve((size_t)DA_ * DQ_ * 2));
  unsigned short* Wq_fus_bf = (unsigned short*)(ws + carve((size_t)GDH_ * DQ_ * 2));
  unsigned short* Wc_bf     = (unsigned short*)(ws + carve((size_t)NC_ * GDH_ * 2));
  unsigned short* Wv_fus_bf = (unsigned short*)(ws + carve((size_t)G_ * DH_ * DV_ * 2));
  unsigned short* xq_bf     = (unsigned short*)(ws + carve((size_t)B_ * DA_ * 2));
  unsigned short* xv_bf     = (unsigned short*)(ws + carve((size_t)MTOT_ * DA_ * 2));
  float*          att_f     = (float*)         (ws + carve((size_t)B_ * G_ * NPOS_ * 4));
  unsigned short* vatt_bf   = (unsigned short*)(ws + carve((size_t)B_ * G_ * DV_ * 2));
  unsigned short* vfus_bf   = (unsigned short*)(ws + carve((size_t)B_ * GDH_ * 2));
  unsigned short* qfus_bf   = (unsigned short*)(ws + carve((size_t)B_ * GDH_ * 2));
  unsigned short* prod_bf   = (unsigned short*)(ws + carve((size_t)B_ * GDH_ * 2));

  auto cvt = [&](const float* src, unsigned short* dst, long long n) {
    int blocks = (int)((n + 255) / 256);
    if (blocks > 65535) blocks = 65535;
    k_f32_to_bf16<<<blocks, 256, 0, stream>>>(src, dst, n);
  };

  // 1) convert inputs/weights to bf16
  {
    long long totalV = (long long)B_ * DV_ * NPOS_;
    int blocks = (int)((totalV + 255) / 256);
    k_transpose_v<<<blocks, 256, 0, stream>>>(input_v, Xv_bf);
  }
  cvt(input_q, q_bf,      (long long)B_ * DQ_);
  cvt(Wv_att,  Wv_att_bf, (long long)DA_ * DV_);
  cvt(Wq_att,  Wq_att_bf, (long long)DA_ * DQ_);
  cvt(Wq_fus,  Wq_fus_bf, (long long)GDH_ * DQ_);
  cvt(Wc,      Wc_bf,     (long long)NC_ * GDH_);
  cvt(Wv_fus,  Wv_fus_bf, (long long)G_ * DH_ * DV_);

  auto gemm = [&](const unsigned short* A, int lda, const unsigned short* W, int ldb,
                  const float* bias, void* C, int ldc, int M, int N, int K,
                  bool do_tanh, bool out_bf) {
    dim3 grid((N + 127) / 128, (M + 127) / 128);
    if (do_tanh && out_bf)
      k_gemm_bf16<true, true><<<grid, 256, 0, stream>>>(A, lda, W, ldb, bias, C, ldc, M, N, K);
    else if (do_tanh && !out_bf)
      k_gemm_bf16<true, false><<<grid, 256, 0, stream>>>(A, lda, W, ldb, bias, C, ldc, M, N, K);
    else if (!do_tanh && out_bf)
      k_gemm_bf16<false, true><<<grid, 256, 0, stream>>>(A, lda, W, ldb, bias, C, ldc, M, N, K);
    else
      k_gemm_bf16<false, false><<<grid, 256, 0, stream>>>(A, lda, W, ldb, bias, C, ldc, M, N, K);
  };

  // 2) attention-branch projections
  gemm(q_bf, DQ_, Wq_att_bf, DQ_, bq_att, xq_bf, DA_, B_, DA_, DQ_,  true, true);    // xq
  gemm(Xv_bf, DV_, Wv_att_bf, DV_, bv_att, xv_bf, DA_, MTOT_, DA_, DV_, true, true); // xv (main GEMM)

  // 3) glimpse scores -> d_out wgt region
  k_wgt<<<(B_ * NPOS_) / 8, 256, 0, stream>>>(xv_bf, xq_bf, Watt, batt, wgt_out);

  // 4) softmax over positions
  k_softmax<<<(B_ * G_ + 7) / 8, 256, 0, stream>>>(wgt_out, att_f);

  // 5) attention-weighted pooling of original features
  k_vatt<<<(B_ * DV_) / 8, 256, 0, stream>>>(input_v, att_f, vatt_bf);

  // 6) glimpse fusion + question fusion
  for (int g = 0; g < G_; ++g) {
    gemm(vatt_bf + (size_t)g * DV_, G_ * DV_,
         Wv_fus_bf + (size_t)g * DH_ * DV_, DV_,
         bv_fus + (size_t)g * DH_,
         vfus_bf + (size_t)g * DH_, GDH_,
         B_, DH_, DV_, true, true);
  }
  gemm(q_bf, DQ_, Wq_fus_bf, DQ_, bq_fus, qfus_bf, GDH_, B_, GDH_, DQ_, true, true);

  // 7) Hadamard fuse
  k_hadamard_bf<<<(B_ * GDH_ + 255) / 256, 256, 0, stream>>>(vfus_bf, qfus_bf, prod_bf, B_ * GDH_);

  // 8) classifier -> d_out x region (f32)
  gemm(prod_bf, GDH_, Wc_bf, GDH_, bc, x_out, NC_, B_, NC_, GDH_, false, false);
}
